// RecursiveMLP_7146825580738
// MI455X (gfx1250) — matrix operand order repaired and checked by reading, not compiled, not run on hardware
//
#include <hip/hip_runtime.h>

typedef __attribute__((ext_vector_type(16))) _Float16 v16h;
typedef __attribute__((ext_vector_type(8)))  _Float16 v8h;
typedef __attribute__((ext_vector_type(8)))  float    v8f;

union F16x16 { v16h v; v8h h2[2]; _Float16 h[16]; unsigned u[8]; };
union F8f    { v8f v; float f[8]; };

#define UNITS 128
#define WSTR  136          // padded LDS row stride (halves) for W1T/W2T (bank decorrelation, 16B multiple)
#define RS    96           // ring buffer length per row (32 seed + 64 outputs)
#define XS    192          // x[:,32:224] slice length per row
#define NSTEP 64
#define ROWS_PER_WAVE  16
#define WAVES          4
#define ROWS_PER_BLOCK (ROWS_PER_WAVE*WAVES)

__global__ __launch_bounds__(128)
void rmlp_wmma_kernel(const float* __restrict__ x,
                      const float* __restrict__ W1, const float* __restrict__ b1,
                      const float* __restrict__ W2, const float* __restrict__ b2,
                      const float* __restrict__ W3, const float* __restrict__ b3p,
                      float* __restrict__ out, int Btot)
{
    extern __shared__ _Float16 smem[];
    _Float16* W1T   = smem;                  // 128 x WSTR halves (W1 transposed, K padded 97->128)
    _Float16* W2T   = W1T + UNITS * WSTR;    // 128 x WSTR halves (W2 transposed)
    _Float16* dbase = W2T + UNITS * WSTR;

    const int tid  = threadIdx.x;
    const int lane = tid & 31;
    const int wave = tid >> 5;
    const int bcol = lane & 15;      // batch column within tile
    const int hi   = lane >> 4;      // lane-half
    const int hi8  = hi << 3;
    const int hi16 = hi << 4;

    // ---- stage W1^T / W2^T into LDS as f16 (coalesced global reads) ----
    for (int id = tid; id < UNITS * UNITS; id += blockDim.x) {
        int n = id & (UNITS - 1);
        int k = id >> 7;
        W1T[n * WSTR + k] = (k < 97) ? (_Float16)W1[k * UNITS + n] : (_Float16)0.0f;
        W2T[n * WSTR + k] = (_Float16)W2[k * UNITS + n];
    }

    // ---- per-wave per-row staging: ring R (f16) and x-slice Xv (f16) ----
    _Float16* Rr = dbase + wave * (ROWS_PER_WAVE * (RS + XS));
    _Float16* Xv = Rr + ROWS_PER_WAVE * RS;
    const long rowg = (long)blockIdx.x * ROWS_PER_BLOCK + wave * ROWS_PER_WAVE;
    for (int id = lane; id < ROWS_PER_WAVE * 224; id += 32) {
        int r = id / 224, c = id - r * 224;
        long gr = rowg + r; if (gr >= Btot) gr = Btot - 1;
        float xv = x[gr * 224 + c];
        if (c < 32) Rr[r * RS + c]        = (_Float16)xv;   // e buffer seed
        else        Xv[r * XS + (c - 32)] = (_Float16)xv;   // v data
    }
    __syncthreads();

    // ---- bias C-init fragments in registers (C layout: M = r + 8*hi, N = bcol) ----
    v8f b1f[8], b2f[8];
#pragma unroll
    for (int mt = 0; mt < 8; ++mt) {
        F8f t1, t2;
#pragma unroll
        for (int r = 0; r < 8; ++r) {
            t1.f[r] = b1[mt * 16 + r + hi8];
            t2.f[r] = b2[mt * 16 + r + hi8];
        }
        b1f[mt] = t1.v; b2f[mt] = t2.v;
    }
    const float b3v = b3p[0];

    // ---- layer-3 A fragments (W3^T as 16x128, only row M==0 non-zero) ----
    v16h a3[4];
#pragma unroll
    for (int kt = 0; kt < 4; ++kt) {
        F16x16 f;
#pragma unroll
        for (int j = 0; j < 16; ++j) f.h[j] = (_Float16)0.0f;
        if (bcol == 0) {                       // lanes 0 and 16 carry row M=0
            int off = kt * 32 + hi16;
#pragma unroll
            for (int j = 0; j < 8; ++j) {
                f.h[j]     = (_Float16)W3[off + j];
                f.h[8 + j] = (_Float16)W3[off + 8 + j];
            }
        }
        a3[kt] = f.v;
    }

    // relu -> f16 -> cross-half exchange: build next-layer B fragments from D tiles
    auto packB = [&](const v8f* dt, v16h* bout) {
#pragma unroll
        for (int kt = 0; kt < 4; ++kt) {
            F8f t0, t1; t0.v = dt[2 * kt]; t1.v = dt[2 * kt + 1];
            F16x16 kf, sf;
#pragma unroll
            for (int j = 0; j < 8; ++j) {
                float r0 = t0.f[j] > 0.f ? t0.f[j] : 0.f;
                float r1 = t1.f[j] > 0.f ? t1.f[j] : 0.f;
                kf.h[j] = (_Float16)(hi ? r1 : r0);   // half this lane keeps
                sf.h[j] = (_Float16)(hi ? r0 : r1);   // half sent to lane^16
            }
            F16x16 fr;
#pragma unroll
            for (int q = 0; q < 4; ++q) {
                unsigned rv = (unsigned)__shfl_xor((int)sf.u[q], 16, 32);
                fr.u[q]     = hi ? rv      : kf.u[q];
                fr.u[4 + q] = hi ? kf.u[q] : rv;
            }
            bout[kt] = fr.v;
        }
    };

    // ================= 64-step recurrence =================
    for (int i = 0; i < NSTEP; ++i) {
        // --- layer-1 B fragments = data^T (K striped: lanes0-15 K=0..15, lanes16-31 K=16..31) ---
        v16h bfrag[4];
        {
            F16x16 f0, f1, f2, f3;
            const _Float16* rrow = Rr + bcol * RS + i + hi16;       // k in [0,32): ring window
#pragma unroll
            for (int j = 0; j < 16; ++j) f0.h[j] = rrow[j];
            const _Float16* xrow = Xv + bcol * XS + 2 * i + hi16;   // k in [32,96): x slice
#pragma unroll
            for (int j = 0; j < 16; ++j) f1.h[j] = xrow[j];
#pragma unroll
            for (int j = 0; j < 16; ++j) f2.h[j] = xrow[32 + j];
#pragma unroll
            for (int j = 0; j < 16; ++j) f3.h[j] = (_Float16)0.0f;  // k in [96,128): pad
            if (hi == 0) f3.h[0] = (_Float16)(float)i;              // k==96 column = step index
            bfrag[0] = f0.v; bfrag[1] = f1.v; bfrag[2] = f2.v; bfrag[3] = f3.v;
        }

        // --- layer 1: D = W1^T x data^T + b1 ---
        v8f d[8];
#pragma unroll
        for (int mt = 0; mt < 8; ++mt) {
            v8f acc = b1f[mt];
            const int rowbase = (mt * 16 + bcol) * WSTR + hi16;
#pragma unroll
            for (int kt = 0; kt < 4; ++kt) {
                F16x16 a;
                a.h2[0] = *(const v8h*)(W1T + rowbase + kt * 32);
                a.h2[1] = *(const v8h*)(W1T + rowbase + kt * 32 + 8);
                acc = __builtin_amdgcn_wmma_f32_16x16x32_f16(false, a.v, false, bfrag[kt],
                                                             (short)0, acc, false, false);
            }
            d[mt] = acc;
        }

        // --- layer 2 ---
        v16h b2frag[4];
        packB(d, b2frag);
#pragma unroll
        for (int mt = 0; mt < 8; ++mt) {
            v8f acc = b2f[mt];
            const int rowbase = (mt * 16 + bcol) * WSTR + hi16;
#pragma unroll
            for (int kt = 0; kt < 4; ++kt) {
                F16x16 a;
                a.h2[0] = *(const v8h*)(W2T + rowbase + kt * 32);
                a.h2[1] = *(const v8h*)(W2T + rowbase + kt * 32 + 8);
                acc = __builtin_amdgcn_wmma_f32_16x16x32_f16(false, a.v, false, b2frag[kt],
                                                             (short)0, acc, false, false);
            }
            d[mt] = acc;
        }

        // --- layer 3: w = W3^T x h2^T + b3 (row 0 of D; result at lane=bcol, vgpr0) ---
        v16h b3frag[4];
        packB(d, b3frag);
        F8f ai; 
#pragma unroll
        for (int q = 0; q < 8; ++q) ai.f[q] = 0.f;
        if (hi == 0) ai.f[0] = b3v;
        v8f acc3 = ai.v;
#pragma unroll
        for (int kt = 0; kt < 4; ++kt)
            acc3 = __builtin_amdgcn_wmma_f32_16x16x32_f16(false, a3[kt], false, b3frag[kt],
                                                          (short)0, acc3, false, false);
        F8f res; res.v = acc3;
        float w = res.f[0];

        // --- epilogue: emit output, append to ring (feeds step i+1) ---
        if (hi == 0) {
            Rr[bcol * RS + 32 + i] = (_Float16)w;
            long gr = rowg + bcol;
            if (gr < Btot) out[gr * (long)NSTEP + i] = w;
        }
    }
}

extern "C" void kernel_launch(void* const* d_in, const int* in_sizes, int n_in,
                              void* d_out, int out_size, void* d_ws, size_t ws_size,
                              hipStream_t stream) {
    const float* x  = (const float*)d_in[0];
    const float* W1 = (const float*)d_in[1];
    const float* b1 = (const float*)d_in[2];
    const float* W2 = (const float*)d_in[3];
    const float* b2 = (const float*)d_in[4];
    const float* W3 = (const float*)d_in[5];
    const float* b3 = (const float*)d_in[6];
    const int Btot = in_sizes[0] / 224;
    const int grid = (Btot + ROWS_PER_BLOCK - 1) / ROWS_PER_BLOCK;
    const size_t shmem = (size_t)(2 * UNITS * WSTR + WAVES * ROWS_PER_WAVE * (RS + XS)) * 2;
    rmlp_wmma_kernel<<<grid, 128, shmem, stream>>>(x, W1, b1, W2, b2, W3, b3,
                                                   (float*)d_out, Btot);
}
